// TextOnlyScanDecoder_88819923682051
// MI455X (gfx1250) — matrix operand order repaired and checked by reading, not compiled, not run on hardware
//
#include <hip/hip_runtime.h>
#include <math.h>

// ---------------- types ----------------
typedef __bf16 bf16_t;
typedef bf16_t bf16x16 __attribute__((ext_vector_type(16)));
typedef bf16_t bf16x8  __attribute__((ext_vector_type(8)));
typedef float  f32x8   __attribute__((ext_vector_type(8)));

// ---------------- problem constants ----------------
static constexpr int B_ = 32, T_ = 512, A_ = 512, E_ = 256, H_ = 320, V_ = 8192;
static constexpr int FH = 4 * H_;          // 1280
static constexpr int KIN = A_ + E_;        // 768
static constexpr int M_ = B_ * T_;         // 16384 rows (m = t*32 + b)
static constexpr int REC_BLOCKS = 20;      // persistent kernel blocks (160 waves)

// ---------------- workspace layout (bytes, all 256-aligned sizes) ----------------
static constexpr size_t SZ_WIN  = (size_t)H_ * KIN * 2;
static constexpr size_t SZ_WIH  = (size_t)FH * H_ * 2;
static constexpr size_t SZ_WHH  = (size_t)FH * H_ * 2;
static constexpr size_t SZ_WOUT = (size_t)V_ * H_ * 2;
static constexpr size_t SZ_EMB  = (size_t)V_ * E_ * 2;
static constexpr size_t SZ_BG   = (size_t)FH * 4;
static constexpr size_t SZ_ABF  = (size_t)M_ * KIN * 2;
static constexpr size_t SZ_XBF  = (size_t)M_ * H_ * 2;
static constexpr size_t SZ_G0   = (size_t)M_ * FH * 4;
static constexpr size_t SZ_HALL = (size_t)M_ * H_ * 2;
static constexpr size_t SZ_H    = (size_t)B_ * H_ * 2;
static constexpr size_t SZ_C    = (size_t)B_ * H_ * 4;
static constexpr size_t SZ_GT   = (size_t)B_ * FH * 4;

static constexpr size_t O_WIN  = 0;
static constexpr size_t O_WIH  = O_WIN  + SZ_WIN;
static constexpr size_t O_WHH  = O_WIH  + SZ_WIH;
static constexpr size_t O_WOUT = O_WHH  + SZ_WHH;
static constexpr size_t O_EMB  = O_WOUT + SZ_WOUT;
static constexpr size_t O_BG   = O_EMB  + SZ_EMB;
static constexpr size_t O_ABF  = O_BG   + SZ_BG;
static constexpr size_t O_XBF  = O_ABF  + SZ_ABF;
static constexpr size_t O_G0   = O_XBF  + SZ_XBF;
static constexpr size_t O_HALL = O_G0   + SZ_G0;
static constexpr size_t O_H    = O_HALL + SZ_HALL;
static constexpr size_t O_C    = O_H    + SZ_H;
static constexpr size_t O_GT   = O_C    + SZ_C;
static constexpr size_t O_BAR  = O_GT   + SZ_GT;

// ---------------- helpers ----------------
__device__ inline bf16_t f2bf(float f) {
  unsigned u = __float_as_uint(f);
  unsigned r = u + 0x7FFFu + ((u >> 16) & 1u);   // round-to-nearest-even
  union { unsigned short s; bf16_t b; } c;
  c.s = (unsigned short)(r >> 16);
  return c.b;
}

__device__ inline float sigm(float x) { return 1.0f / (1.0f + __expf(-x)); }

__device__ inline bf16x16 load_frag(const bf16_t* p0, const bf16_t* p1) {
  union { bf16x16 v; bf16x8 h[2]; } u;
  u.h[0] = *reinterpret_cast<const bf16x8*>(p0);
  u.h[1] = *reinterpret_cast<const bf16x8*>(p1);
  return u.v;
}

// ---------------- kernel: convert weights to bf16, fuse biases ----------------
static constexpr size_t N_EMB = (size_t)V_ * E_;
static constexpr size_t N_WIN = (size_t)H_ * KIN;
static constexpr size_t N_WIH = (size_t)FH * H_;
static constexpr size_t N_WHH = (size_t)FH * H_;
static constexpr size_t N_WOU = (size_t)V_ * H_;
static constexpr size_t N_PREP = N_EMB + N_WIN + N_WIH + N_WHH + N_WOU + FH;

__global__ void prep_weights(const float* __restrict__ emb,
                             const float* __restrict__ W_in,
                             const float* __restrict__ W_ih,
                             const float* __restrict__ W_hh,
                             const float* __restrict__ W_out,
                             const float* __restrict__ b_ih,
                             const float* __restrict__ b_hh,
                             bf16_t* __restrict__ emb_bf,
                             bf16_t* __restrict__ Win_bf,
                             bf16_t* __restrict__ Wih_bf,
                             bf16_t* __restrict__ Whh_bf,
                             bf16_t* __restrict__ Wout_bf,
                             float* __restrict__ bg) {
  size_t i = (size_t)blockIdx.x * blockDim.x + threadIdx.x;
  if (i >= N_PREP) return;
  size_t k = i;
  if (k < N_EMB) { emb_bf[k] = f2bf(k < (size_t)E_ ? 0.0f : emb[k]); return; } // pad row 0
  k -= N_EMB;
  if (k < N_WIN) { Win_bf[k] = f2bf(W_in[k]); return; }
  k -= N_WIN;
  if (k < N_WIH) { Wih_bf[k] = f2bf(W_ih[k]); return; }
  k -= N_WIH;
  if (k < N_WHH) { Whh_bf[k] = f2bf(W_hh[k]); return; }
  k -= N_WHH;
  if (k < N_WOU) { Wout_bf[k] = f2bf(W_out[k]); return; }
  k -= N_WOU;
  bg[k] = b_ih[k] + b_hh[k];
}

// ---------------- kernel: build concat(acoustic, emb[tok]) as bf16 [M, 768] ----------------
__global__ void build_A(const float* __restrict__ ac, const int* __restrict__ tok,
                        const bf16_t* __restrict__ emb_bf, bf16_t* __restrict__ Abf) {
  size_t i = (size_t)blockIdx.x * blockDim.x + threadIdx.x;
  if (i >= (size_t)M_ * KIN) return;
  size_t m = i / KIN;
  int k = (int)(i % KIN);
  int b = (int)(m & 31), t = (int)(m >> 5);
  bf16_t v;
  if (k < A_) v = f2bf(ac[((size_t)b * T_ + t) * A_ + k]);
  else        v = emb_bf[(size_t)tok[b * T_ + t] * E_ + (k - A_)];
  Abf[i] = v;
}

// ---------------- WMMA GEMM: C[M,N] = act(A[M,K] @ Bw[N,K]^T + bias[N]) ----------------
// each wave computes a 32x64 tile (8 accumulators), K stepped by 32;
// per K-chunk: 2 A-frags + 4 B-frags feed 8 WMMAs (loads/WMMA = 0.75)
template <int ACT, int OUTBF, int REMAP>
__global__ __launch_bounds__(256) void gemm_bf16(const bf16_t* __restrict__ A, int lda,
                                                 const bf16_t* __restrict__ Bw, int ldb,
                                                 const float* __restrict__ bias,
                                                 float* __restrict__ Cf,
                                                 bf16_t* __restrict__ Cb, int ldc,
                                                 int K, int ntiles_n) {
  int wave = (int)((blockIdx.x * blockDim.x + threadIdx.x) >> 5);
  int lane = (int)(threadIdx.x & 31);
  int tile_m = wave / ntiles_n;
  int tile_n = wave % ntiles_n;
  int m0 = tile_m * 32, n0 = tile_n * 64;
  int al = lane & 15, ah = lane >> 4;

  f32x8 acc[2][4] = {};
  const bf16_t* arow0 = A + (size_t)(m0 + al) * lda + ah * 16;
  const bf16_t* arow1 = A + (size_t)(m0 + 16 + al) * lda + ah * 16;
  const bf16_t* brow0 = Bw + (size_t)(n0 + 0 * 16 + al) * ldb + ah * 16;
  const bf16_t* brow1 = Bw + (size_t)(n0 + 1 * 16 + al) * ldb + ah * 16;
  const bf16_t* brow2 = Bw + (size_t)(n0 + 2 * 16 + al) * ldb + ah * 16;
  const bf16_t* brow3 = Bw + (size_t)(n0 + 3 * 16 + al) * ldb + ah * 16;

  for (int k0 = 0; k0 < K; k0 += 32) {
    bf16x16 a0 = load_frag(arow0 + k0, arow0 + k0 + 8);
    bf16x16 a1 = load_frag(arow1 + k0, arow1 + k0 + 8);
    bf16x16 b0 = load_frag(brow0 + k0, brow0 + k0 + 8);
    bf16x16 b1 = load_frag(brow1 + k0, brow1 + k0 + 8);
    bf16x16 b2 = load_frag(brow2 + k0, brow2 + k0 + 8);
    bf16x16 b3 = load_frag(brow3 + k0, brow3 + k0 + 8);
    acc[0][0] = __builtin_amdgcn_wmma_f32_16x16x32_bf16(false, a0, false, b0, (short)0, acc[0][0], false, false);
    acc[1][0] = __builtin_amdgcn_wmma_f32_16x16x32_bf16(false, a1, false, b0, (short)0, acc[1][0], false, false);
    acc[0][1] = __builtin_amdgcn_wmma_f32_16x16x32_bf16(false, a0, false, b1, (short)0, acc[0][1], false, false);
    acc[1][1] = __builtin_amdgcn_wmma_f32_16x16x32_bf16(false, a1, false, b1, (short)0, acc[1][1], false, false);
    acc[0][2] = __builtin_amdgcn_wmma_f32_16x16x32_bf16(false, a0, false, b2, (short)0, acc[0][2], false, false);
    acc[1][2] = __builtin_amdgcn_wmma_f32_16x16x32_bf16(false, a1, false, b2, (short)0, acc[1][2], false, false);
    acc[0][3] = __builtin_amdgcn_wmma_f32_16x16x32_bf16(false, a0, false, b3, (short)0, acc[0][3], false, false);
    acc[1][3] = __builtin_amdgcn_wmma_f32_16x16x32_bf16(false, a1, false, b3, (short)0, acc[1][3], false, false);
  }

#pragma unroll
  for (int i = 0; i < 2; ++i) {
#pragma unroll
    for (int j = 0; j < 4; ++j) {
      int n = n0 + j * 16 + al;
      float bn = bias[n];
#pragma unroll
      for (int r = 0; r < 8; ++r) {
        int m = m0 + i * 16 + ah * 8 + r;
        float v = acc[i][j][r] + bn;
        if (ACT == 1) v = tanhf(v);
        size_t row = REMAP ? ((size_t)(m & 31) * T_ + (size_t)(m >> 5)) : (size_t)m;
        if (OUTBF) Cb[row * ldc + n] = f2bf(v);
        else       Cf[row * ldc + n] = v;
      }
    }
  }
}

// ---------------- init state + barrier counter ----------------
__global__ void init_state(bf16_t* __restrict__ h, float* __restrict__ c, unsigned* __restrict__ bar) {
  int i = blockIdx.x * blockDim.x + threadIdx.x;
  if (i == 0) *bar = 0u;
  if (i < B_ * H_) {
    union { unsigned short s; bf16_t b; } z; z.s = 0;
    h[i] = z.b;
    c[i] = 0.0f;
  }
}

// ---------------- grid barrier (persistent kernel) ----------------
__device__ inline void grid_bar(unsigned* cnt, unsigned nblocks, unsigned* bars) {
  __threadfence();
  __syncthreads();
  if (threadIdx.x == 0) {
    atomicAdd(cnt, 1u);
    unsigned target = (++(*bars)) * nblocks;
    while (atomicAdd(cnt, 0u) < target) __builtin_amdgcn_s_sleep(2);
  } else {
    ++(*bars);
  }
  __syncthreads();
  __threadfence();
}

// ---------------- persistent LSTM recurrence ----------------
// 20 blocks x 256 threads = 160 waves; each wave owns one 16x16 gate tile.
// W_hh fragments are preloaded into registers once; per step all 10 h-frags
// are loaded with one wait, and the K reduction uses 2 accumulators to break
// the WMMA->WMMA RAW chain.
__global__ __launch_bounds__(256) void lstm_recurrence(const float* __restrict__ G0,
                                                       const bf16_t* __restrict__ Whh,
                                                       bf16_t* __restrict__ h_cur,
                                                       float* __restrict__ c_cur,
                                                       float* __restrict__ gates,
                                                       bf16_t* __restrict__ Hall,
                                                       unsigned* __restrict__ bar) {
  int wave = (int)((blockIdx.x * blockDim.x + threadIdx.x) >> 5);
  int lane = (int)(threadIdx.x & 31);
  int tile_n = wave >> 1;          // 0..79 over 4H
  int tile_m = wave & 1;           // 0..1 over B
  int al = lane & 15, ah = lane >> 4;
  int n = tile_n * 16 + al;
  const bf16_t* brow = Whh + (size_t)n * H_ + ah * 16;
  const bf16_t* arow = h_cur + (size_t)(tile_m * 16 + al) * H_ + ah * 16;
  unsigned bars = 0;

  // W_hh fragments resident in VGPRs for all 512 steps
  bf16x16 bfrag[10];
#pragma unroll
  for (int k = 0; k < 10; ++k)
    bfrag[k] = load_frag(brow + k * 32, brow + k * 32 + 8);

  for (int t = 0; t < T_; ++t) {
    // load entire h-row slice for this lane (10 frags, one wait)
    bf16x16 af[10];
#pragma unroll
    for (int k = 0; k < 10; ++k)
      af[k] = load_frag(arow + k * 32, arow + k * 32 + 8);

    // C-init from precomputed pre-gates (already includes b_ih + b_hh)
    f32x8 acc0, acc1 = {};
    const float* g0p = G0 + ((size_t)t * B_ + tile_m * 16 + ah * 8) * FH + n;
#pragma unroll
    for (int r = 0; r < 8; ++r) acc0[r] = g0p[(size_t)r * FH];

#pragma unroll
    for (int k = 0; k < 10; k += 2) {
      acc0 = __builtin_amdgcn_wmma_f32_16x16x32_bf16(false, af[k],     false, bfrag[k],     (short)0, acc0, false, false);
      acc1 = __builtin_amdgcn_wmma_f32_16x16x32_bf16(false, af[k + 1], false, bfrag[k + 1], (short)0, acc1, false, false);
    }

    float* gp = gates + (size_t)(tile_m * 16 + ah * 8) * FH + n;
#pragma unroll
    for (int r = 0; r < 8; ++r) gp[(size_t)r * FH] = acc0[r] + acc1[r];

    grid_bar(bar, REC_BLOCKS, &bars);

    // pointwise LSTM: 32*320 = 10240 elems over 5120 threads (exactly 2 each)
    for (int idx = blockIdx.x * 256 + (int)threadIdx.x; idx < B_ * H_; idx += REC_BLOCKS * 256) {
      int b = idx / H_, j = idx % H_;
      const float* g = gates + (size_t)b * FH;
      float ig = g[j], fg = g[H_ + j], gg = g[2 * H_ + j], og = g[3 * H_ + j];
      float cn = sigm(fg) * c_cur[idx] + sigm(ig) * tanhf(gg);
      float hn = sigm(og) * tanhf(cn);
      c_cur[idx] = cn;
      bf16_t hb = f2bf(hn);
      h_cur[idx] = hb;
      Hall[((size_t)t * B_ + b) * H_ + j] = hb;
    }

    grid_bar(bar, REC_BLOCKS, &bars);
  }
}

// ---------------- in-place log_softmax over rows of 8192 ----------------
__global__ __launch_bounds__(256) void logsoftmax_rows(float* __restrict__ out) {
  __shared__ float red[256];
  float* p = out + (size_t)blockIdx.x * V_;
  int tid = threadIdx.x;

  float mx = -INFINITY;
  for (int i = tid; i < V_; i += 256) mx = fmaxf(mx, p[i]);
  red[tid] = mx; __syncthreads();
  for (int s = 128; s > 0; s >>= 1) {
    if (tid < s) red[tid] = fmaxf(red[tid], red[tid + s]);
    __syncthreads();
  }
  mx = red[0]; __syncthreads();

  float sum = 0.0f;
  for (int i = tid; i < V_; i += 256) sum += __expf(p[i] - mx);
  red[tid] = sum; __syncthreads();
  for (int s = 128; s > 0; s >>= 1) {
    if (tid < s) red[tid] += red[tid + s];
    __syncthreads();
  }
  float lse = mx + __logf(red[0]);
  __syncthreads();

  for (int i = tid; i < V_; i += 256) p[i] -= lse;
}

// ---------------- host launch ----------------
extern "C" void kernel_launch(void* const* d_in, const int* in_sizes, int n_in,
                              void* d_out, int out_size, void* d_ws, size_t ws_size,
                              hipStream_t stream) {
  const float* acoustic = (const float*)d_in[0];
  const int*   dec_in   = (const int*)d_in[1];
  const float* emb      = (const float*)d_in[2];
  const float* W_in     = (const float*)d_in[3];
  const float* b_in     = (const float*)d_in[4];
  const float* W_ih     = (const float*)d_in[5];
  const float* W_hh     = (const float*)d_in[6];
  const float* b_ih     = (const float*)d_in[7];
  const float* b_hh     = (const float*)d_in[8];
  const float* W_out    = (const float*)d_in[9];
  const float* b_out    = (const float*)d_in[10];
  float* out = (float*)d_out;

  char* ws = (char*)d_ws;
  bf16_t* Win_bf  = (bf16_t*)(ws + O_WIN);
  bf16_t* Wih_bf  = (bf16_t*)(ws + O_WIH);
  bf16_t* Whh_bf  = (bf16_t*)(ws + O_WHH);
  bf16_t* Wout_bf = (bf16_t*)(ws + O_WOUT);
  bf16_t* emb_bf  = (bf16_t*)(ws + O_EMB);
  float*  bg      = (float*) (ws + O_BG);
  bf16_t* Abf     = (bf16_t*)(ws + O_ABF);
  bf16_t* Xbf     = (bf16_t*)(ws + O_XBF);
  float*  G0      = (float*) (ws + O_G0);
  bf16_t* Hall    = (bf16_t*)(ws + O_HALL);
  bf16_t* h_cur   = (bf16_t*)(ws + O_H);
  float*  c_cur   = (float*) (ws + O_C);
  float*  gates   = (float*) (ws + O_GT);
  unsigned* bar   = (unsigned*)(ws + O_BAR);

  // 1) weights -> bf16, fused bias
  {
    int blocks = (int)((N_PREP + 255) / 256);
    prep_weights<<<blocks, 256, 0, stream>>>(emb, W_in, W_ih, W_hh, W_out, b_ih, b_hh,
                                             emb_bf, Win_bf, Wih_bf, Whh_bf, Wout_bf, bg);
  }
  // 2) build concat input [16384, 768] bf16
  {
    size_t tot = (size_t)M_ * KIN;
    build_A<<<(int)((tot + 255) / 256), 256, 0, stream>>>(acoustic, dec_in, emb_bf, Abf);
  }
  // 3) X = tanh(Abf @ W_in^T + b_in)  -> bf16 [16384, 320]
  gemm_bf16<1, 1, 0><<<(M_ / 32) * (H_ / 64) / 8, 256, 0, stream>>>(
      Abf, KIN, Win_bf, KIN, b_in, nullptr, Xbf, H_, KIN, H_ / 64);
  // 4) G0 = X @ W_ih^T + (b_ih + b_hh)  -> f32 [16384, 1280]
  gemm_bf16<0, 0, 0><<<(M_ / 32) * (FH / 64) / 8, 256, 0, stream>>>(
      Xbf, H_, Wih_bf, H_, bg, G0, nullptr, FH, H_, FH / 64);
  // 5) zero h, c, barrier counter
  init_state<<<(B_ * H_ + 255) / 256, 256, 0, stream>>>(h_cur, c_cur, bar);
  // 6) sequential LSTM recurrence (persistent, grid-barriered)
  lstm_recurrence<<<REC_BLOCKS, 256, 0, stream>>>(G0, Whh_bf, h_cur, c_cur, gates, Hall, bar);
  // 7) logits = Hall @ W_out^T + b_out -> d_out [B,T,V] (row remap m=t*32+b -> b*512+t)
  gemm_bf16<0, 0, 1><<<(M_ / 32) * (V_ / 64) / 8, 256, 0, stream>>>(
      Hall, H_, Wout_bf, H_, b_out, out, nullptr, V_, H_, V_ / 64);
  // 8) in-place log_softmax per (b,t) row
  logsoftmax_rows<<<M_, 256, 0, stream>>>(out);
}
